// Aggregator_67010079752193
// MI455X (gfx1250) — compile-verified, matrix operand order checked by reading, not run on hardware
//
#include <hip/hip_runtime.h>

#define N_NODES 100000
#define N_EDGES 600000
#define D_FEAT 128
#define OUT_DIM 128

typedef __attribute__((ext_vector_type(16))) __bf16 v16bf;
typedef __attribute__((ext_vector_type(8)))  float  v8f;

__device__ __forceinline__ unsigned short f2bf_bits(float f) {
  __bf16 b = (__bf16)f;
  return __builtin_bit_cast(unsigned short, b);
}

// ---------------------------------------------------------------------------
// Kernel 0: zero the h accumulator region of the workspace (float4 stores)
// ---------------------------------------------------------------------------
__global__ void zero_h_kernel(float4* h4, int n4) {
  int i = blockIdx.x * blockDim.x + threadIdx.x;
  if (i < n4) h4[i] = make_float4(0.f, 0.f, 0.f, 0.f);
}

// ---------------------------------------------------------------------------
// Kernel 1: pre-swizzle W (256x128 f32) into bf16 B-matrix fragments matching
// the CDNA5 16x16x32 bf16 WMMA lane layout.
//   Fragment index: ((ntile*8 + kchunk)*32 + lane), 16 bf16 elements each.
//   lane<16  -> this lane's K offsets are {kbase+0..7, kbase+16..23}
//   lane>=16 -> this lane's K offsets are {kbase+8..15, kbase+24..31}
//   column n = ntile*16 + (lane & 15)
// ---------------------------------------------------------------------------
__global__ void prep_wfrag_kernel(const float* __restrict__ W,
                                  unsigned short* __restrict__ wf) {
  int t = blockIdx.x * blockDim.x + threadIdx.x;   // 0 .. 8*8*32-1
  if (t >= 8 * 8 * 32) return;
  int lane  = t & 31;
  int kc    = (t >> 5) & 7;    // k-chunk (0..7 over combined K=256)
  int ntile = t >> 8;          // output column tile (0..7)
  int n     = ntile * 16 + (lane & 15);
  int koff  = (lane < 16) ? 0 : 8;
  int kbase = kc * 32;
  unsigned short* dst = wf + t * 16;
#pragma unroll
  for (int i = 0; i < 8; ++i) {
    int k0 = kbase + koff + i;        // low K group
    int k1 = kbase + 16 + koff + i;   // high K group
    dst[i]     = f2bf_bits(W[k0 * OUT_DIM + n]);
    dst[8 + i] = f2bf_bits(W[k1 * OUT_DIM + n]);
  }
}

// ---------------------------------------------------------------------------
// Kernel 2: COO SpMM scatter: h[row] += val * x[col]
// One wave (32 lanes) per edge; lane L handles features 4L..4L+3.
// Edge metadata is wave-uniform -> scalar (SMEM) loads; x row read is a
// coalesced 512B burst (L2-resident: x is 51.2MB < 192MB L2); accumulation
// via non-returning global_atomic_add_f32 at the L2 atomic units.
// ---------------------------------------------------------------------------
__global__ void spmm_atomic_kernel(const float* __restrict__ x,
                                   const int* __restrict__ rows,
                                   const int* __restrict__ cols,
                                   const float* __restrict__ vals,
                                   float* __restrict__ h) {
  int wave = blockIdx.x * (blockDim.x >> 5) + (threadIdx.x >> 5);
  int lane = threadIdx.x & 31;
  if (wave >= N_EDGES) return;
  int   row = rows[wave];
  int   col = cols[wave];
  float val = vals[wave];
  const float4 xv = *(const float4*)(x + (size_t)col * D_FEAT + lane * 4);
  float* dst = h + (size_t)row * D_FEAT + lane * 4;
  atomicAdd(dst + 0, val * xv.x);
  atomicAdd(dst + 1, val * xv.y);
  atomicAdd(dst + 2, val * xv.z);
  atomicAdd(dst + 3, val * xv.w);
}

// ---------------------------------------------------------------------------
// Kernel 3: out = relu(h @ W[:128] + x @ W[128:]) via bf16 WMMA, f32 accum.
// Block = 256 threads = 8 waves; block b owns 16 rows (m0 = 16*b),
// wave w owns output column tile w (16 cols). K = 256 combined (h then x),
// consumed as 8 chunks of 32 with v_wmma_f32_16x16x32_bf16.
//
// A-slab staging: the 16x256 f32 slab is shared by all 8 waves, so it is
// loaded from global and converted to bf16 EXACTLY ONCE per block. Wave w
// stages k-chunk w directly in the WMMA A-fragment lane layout into LDS
// (8 KB), then after one barrier every wave consumes fragments with two
// ds_load_b128 per chunk. This removes the 8x redundant global loads and
// 8x redundant f32->bf16 conversions of the naive version, leaving the
// kernel limited by its unique HBM traffic (~24 KB/block).
// ---------------------------------------------------------------------------
__global__ void __launch_bounds__(256)
gemm_wmma_kernel(const float* __restrict__ h,
                 const float* __restrict__ x,
                 const unsigned short* __restrict__ wf,
                 float* __restrict__ out) {
  __shared__ __align__(32) unsigned short ldsA[8 * 32 * 16];  // 8 KB

  const int lane = threadIdx.x & 31;
  const int wv   = threadIdx.x >> 5;      // wave id: stages chunk wv, owns col tile wv
  const int m0   = blockIdx.x * 16;
  const int row  = m0 + (lane & 15);
  const int koff = (lane < 16) ? 0 : 8;

  // ---- Stage: wave wv builds the A fragment for k-chunk kc = wv ----
  {
    const int    kc    = wv;
    const float* src   = (kc < 4) ? h : x;
    const int    kbase = (kc & 3) * 32;
    const float* p = src + (size_t)row * D_FEAT + kbase + koff;
    const float4 f0 = *(const float4*)(p + 0);
    const float4 f1 = *(const float4*)(p + 4);
    const float4 f2 = *(const float4*)(p + 16);
    const float4 f3 = *(const float4*)(p + 20);
    v16bf a;
    a[0]  = (__bf16)f0.x; a[1]  = (__bf16)f0.y;
    a[2]  = (__bf16)f0.z; a[3]  = (__bf16)f0.w;
    a[4]  = (__bf16)f1.x; a[5]  = (__bf16)f1.y;
    a[6]  = (__bf16)f1.z; a[7]  = (__bf16)f1.w;
    a[8]  = (__bf16)f2.x; a[9]  = (__bf16)f2.y;
    a[10] = (__bf16)f2.z; a[11] = (__bf16)f2.w;
    a[12] = (__bf16)f3.x; a[13] = (__bf16)f3.y;
    a[14] = (__bf16)f3.z; a[15] = (__bf16)f3.w;
    *(v16bf*)&ldsA[(kc * 32 + lane) * 16] = a;
  }
  __syncthreads();

  // ---- Compute: 8 chained WMMAs over the combined K=256 ----
  const v16bf* bfrag = (const v16bf*)wf;
  v8f acc = {};
#pragma unroll
  for (int kc = 0; kc < 8; ++kc) {
    const v16bf a = *(const v16bf*)&ldsA[(kc * 32 + lane) * 16];
    const v16bf b = bfrag[(wv * 8 + kc) * 32 + lane];
    acc = __builtin_amdgcn_wmma_f32_16x16x32_bf16(
        /*neg_a=*/false, a, /*neg_b=*/false, b,
        /*c_mod=*/(short)0, acc, /*reuse_a=*/false, /*reuse_b=*/false);
  }

  // C/D layout: vgpr r -> M = r (lanes 0-15) or 8+r (lanes 16-31); N = lane&15
  const int n    = wv * 16 + (lane & 15);
  const int mrow = m0 + ((lane < 16) ? 0 : 8);
#pragma unroll
  for (int r = 0; r < 8; ++r) {
    float v = acc[r];
    out[(size_t)(mrow + r) * OUT_DIM + n] = v > 0.f ? v : 0.f;
  }
}

// ---------------------------------------------------------------------------
// Launch
// ---------------------------------------------------------------------------
extern "C" void kernel_launch(void* const* d_in, const int* in_sizes, int n_in,
                              void* d_out, int out_size, void* d_ws, size_t ws_size,
                              hipStream_t stream) {
  const float* x    = (const float*)d_in[0];
  const int*   rows = (const int*)d_in[1];
  const int*   cols = (const int*)d_in[2];
  const float* vals = (const float*)d_in[3];
  const float* W    = (const float*)d_in[4];
  float*       out  = (float*)d_out;

  // Workspace: [h: N_NODES*D_FEAT f32][wfrag: 8*8*32*16 bf16 = 64KB]
  float*          h  = (float*)d_ws;
  unsigned short* wf = (unsigned short*)((char*)d_ws +
                         (size_t)N_NODES * D_FEAT * sizeof(float));

  // 0) zero h
  {
    const int n4 = (N_NODES * D_FEAT) / 4;
    zero_h_kernel<<<(n4 + 255) / 256, 256, 0, stream>>>((float4*)h, n4);
  }
  // 1) pre-swizzle W into bf16 B fragments
  prep_wfrag_kernel<<<(8 * 8 * 32 + 255) / 256, 256, 0, stream>>>(W, wf);
  // 2) SpMM scatter-add: one wave per edge
  {
    const int waves_per_block = 256 / 32;
    const int blocks = (N_EDGES + waves_per_block - 1) / waves_per_block;
    spmm_atomic_kernel<<<blocks, 256, 0, stream>>>(x, rows, cols, vals, h);
  }
  // 3) fused GEMM + ReLU via bf16 WMMA (LDS-staged A fragments)
  gemm_wmma_kernel<<<N_NODES / 16, 256, 0, stream>>>(h, x, wf, out);
}
